// Net_37890201485673
// MI455X (gfx1250) — compile-verified, matrix operand order checked by reading, not run on hardware
//
#include <hip/hip_runtime.h>
#include <stdint.h>

#define NN 50000
#define NE 400000
#define EPS 1e-5f
#define LDST 40   // padded LDS row stride (ushorts): 32 data + 8 pad, 80B rows (16B aligned)

typedef __attribute__((ext_vector_type(16))) __bf16 v16bf;
typedef __attribute__((ext_vector_type(8)))  float  v8f;

// int4 vector pointee type matching the async-builtin's parameter
typedef int vi4 __attribute__((vector_size(16)));
typedef __attribute__((address_space(1))) vi4* gvi4p;  // global
typedef __attribute__((address_space(3))) vi4* lvi4p;  // LDS

__device__ __forceinline__ unsigned short f2bf(float f) {
    unsigned int u = __float_as_uint(f);
    u = u + 0x7FFFu + ((u >> 16) & 1u);   // round-to-nearest-even
    return (unsigned short)(u >> 16);
}
__device__ __forceinline__ float bf2f(unsigned short h) {
    return __uint_as_float(((unsigned int)h) << 16);
}

// ---- CDNA5 async global->LDS copy (16B per lane), with sync fallback ----
__device__ __forceinline__ void async16(const unsigned short* g, unsigned short* l) {
#if __has_builtin(__builtin_amdgcn_global_load_async_to_lds_b128)
    __builtin_amdgcn_global_load_async_to_lds_b128((gvi4p)(void*)g, (lvi4p)(void*)l, 0, 0);
#else
    *(uint4*)l = *(const uint4*)g;
#endif
}
__device__ __forceinline__ void wait_async0() {
#if __has_builtin(__builtin_amdgcn_s_wait_asynccnt)
    __builtin_amdgcn_s_wait_asynccnt(0);
#else
    asm volatile("s_wait_asynccnt 0x0" ::: "memory");
#endif
}

// ---------------- utility kernels ----------------

__global__ void k_zero(float* __restrict__ p, long long n) {
    long long i = (long long)blockIdx.x * blockDim.x + threadIdx.x;
    long long stride = (long long)gridDim.x * blockDim.x;
    for (; i < n; i += stride) p[i] = 0.0f;
}

__global__ void k_degree(const long long* __restrict__ dst, float* __restrict__ deg) {
    int e = blockIdx.x * blockDim.x + threadIdx.x;
    if (e < NE) atomicAdd(&deg[dst[e]], 1.0f);
}

__global__ void k_cvt_f2bf(const float* __restrict__ x, unsigned short* __restrict__ o, long long n) {
    long long i = (long long)blockIdx.x * blockDim.x + threadIdx.x;
    if (i < n) o[i] = f2bf(x[i]);
}

// W [din, dout] f32  ->  Wt [dout, din] bf16 (transposed so B-frags are contiguous in K)
__global__ void k_cvt_w(const float* __restrict__ W, unsigned short* __restrict__ Wt,
                        int din, int dout) {
    long long i = (long long)blockIdx.x * blockDim.x + threadIdx.x;
    long long n = (long long)din * dout;
    if (i >= n) return;
    int k = (int)(i / dout);
    int c = (int)(i % dout);
    Wt[(long long)c * din + k] = f2bf(W[i]);
}

// mean-aggregation scatter: agg[dst] += h[src] (bf16 -> f32 atomics), 8 feats/thread
__global__ __launch_bounds__(256) void k_scatter(const unsigned short* __restrict__ h,
                                                 const long long* __restrict__ src,
                                                 const long long* __restrict__ dst,
                                                 float* __restrict__ agg, int d) {
    long long idx = (long long)blockIdx.x * blockDim.x + threadIdx.x;
    int per = d >> 3;
    long long total = (long long)NE * per;
    if (idx >= total) return;
    int e = (int)(idx / per);
    int j = (int)(idx % per) << 3;
    long long s = src[e], t = dst[e];
    uint4 v = *(const uint4*)(h + s * (long long)d + j);
    const unsigned short* u = (const unsigned short*)&v;
    float* ap = agg + t * (long long)d + j;
#pragma unroll
    for (int q = 0; q < 8; ++q) atomicAdd(ap + q, bf2f(u[q]));
}

// agg/deg -> bf16
__global__ void k_norm(const float* __restrict__ agg, const float* __restrict__ deg,
                       unsigned short* __restrict__ aggb, int d) {
    long long i = (long long)blockIdx.x * blockDim.x + threadIdx.x;
    long long n = (long long)NN * d;
    if (i >= n) return;
    float dg = deg[i / d];
    dg = dg < 1.0f ? 1.0f : dg;
    aggb[i] = f2bf(agg[i] / dg);
}

// ---------------- fused dual-GEMM: Y = relu?(A1@Wl + A2@Wr + bias) ----------------
// A1/A2: [n, din] bf16 row-major.  B1t/B2t: [dout, din] bf16 (pre-transposed weights).
// Block tile 128x128, BK=32. 8 waves (4x2); wave tile 32x64 = 2x4 wmma accumulators.
// Double-buffered LDS with async global->LDS copies: prefetch tile k+1 during tile k.

__global__ __launch_bounds__(256) void k_gemm(const unsigned short* __restrict__ A1,
                                              const unsigned short* __restrict__ B1t,
                                              const unsigned short* __restrict__ A2,
                                              const unsigned short* __restrict__ B2t,
                                              const float* __restrict__ bias,
                                              float* __restrict__ Y,
                                              int n, int din, int dout, int do_relu) {
    __shared__ __align__(16) unsigned short As[2][128][LDST];
    __shared__ __align__(16) unsigned short Bs[2][128][LDST];

    const int tid  = threadIdx.x;
    const int lane = tid & 31;
    const int wave = tid >> 5;
    const int wm = wave & 3;   // 4 row-groups of 32
    const int wn = wave >> 2;  // 2 col-groups of 64

    const long long rowBase = (long long)blockIdx.x * 128;
    const int colBase = blockIdx.y * 128;

    v8f acc[2][4];
#pragma unroll
    for (int i = 0; i < 2; ++i)
#pragma unroll
        for (int j = 0; j < 4; ++j)
#pragma unroll
            for (int q = 0; q < 8; ++q) acc[i][j][q] = 0.0f;

    // tile-load mapping: 2 threads per row, 16 ushorts (32B = 2 x b128) each
    const int ldRow  = tid >> 1;
    const int ldHalf = (tid & 1) * 16;
    long long aRow = rowBase + ldRow;
    if (aRow >= n) aRow = n - 1;                 // clamped rows never stored
    const int bRowG  = colBase + ldRow;
    const int bValid = bRowG < dout;             // invalid B cols never stored either
    const long long bRow = bValid ? bRowG : 0;

    const int arow0 = wm * 32 + (lane & 15);
    const int kc    = (lane >> 4) * 8;           // ISA 16-bit A/B K-chunk select

    const int steps = din >> 5;                  // K-steps per pass (din % 32 == 0)
    const int ktot  = steps * 2;                 // pass 0: agg@Wl, pass 1: h@Wr

    auto issue_tile = [&](int kt, int buf) {
        const int pass = (kt >= steps) ? 1 : 0;
        const int k0 = (kt - pass * steps) << 5;
        const unsigned short* __restrict__ A  = pass ? A2 : A1;
        const unsigned short* __restrict__ Bt = pass ? B2t : B1t;
        const unsigned short* ga = A + aRow * din + k0 + ldHalf;
        unsigned short* la = &As[buf][ldRow][ldHalf];
        async16(ga, la);
        async16(ga + 8, la + 8);
        if (bValid) {
            const unsigned short* gb = Bt + bRow * din + k0 + ldHalf;
            unsigned short* lb = &Bs[buf][ldRow][ldHalf];
            async16(gb, lb);
            async16(gb + 8, lb + 8);
        }
    };

    issue_tile(0, 0);
    wait_async0();
    __syncthreads();

    for (int kt = 0; kt < ktot; ++kt) {
        const int buf = kt & 1;
        if (kt + 1 < ktot) issue_tile(kt + 1, buf ^ 1);

        // preload all fragments, then 8 back-to-back WMMAs
        const unsigned short (*Asb)[LDST] = As[buf];
        const unsigned short (*Bsb)[LDST] = Bs[buf];
        v16bf afrag[2], bfrag[4];
#pragma unroll
        for (int i = 0; i < 2; ++i) {
            const unsigned short* pa = &Asb[arow0 + i * 16][kc];
            ((uint4*)&afrag[i])[0] = *(const uint4*)pa;
            ((uint4*)&afrag[i])[1] = *(const uint4*)(pa + 16);
        }
#pragma unroll
        for (int j = 0; j < 4; ++j) {
            const unsigned short* pb = &Bsb[wn * 64 + j * 16 + (lane & 15)][kc];
            ((uint4*)&bfrag[j])[0] = *(const uint4*)pb;
            ((uint4*)&bfrag[j])[1] = *(const uint4*)(pb + 16);
        }
#pragma unroll
        for (int j = 0; j < 4; ++j)
#pragma unroll
            for (int i = 0; i < 2; ++i)
                acc[i][j] = __builtin_amdgcn_wmma_f32_16x16x32_bf16(
                    false, afrag[i], false, bfrag[j], (short)0, acc[i][j], false, false);

        wait_async0();        // next tile fully landed in LDS
        __syncthreads();      // all waves done reading buf, next tile visible
    }

    // epilogue: bias (+ReLU), f32 store. C layout: VGPR q -> row q + (lane/16)*8, col lane%16
    const int rsel = (lane >> 4) * 8;
    const int cloc = lane & 15;
#pragma unroll
    for (int j = 0; j < 4; ++j) {
        const int colg = colBase + wn * 64 + j * 16 + cloc;
        if (colg >= dout) continue;
        const float bv = bias[colg];
#pragma unroll
        for (int i = 0; i < 2; ++i) {
#pragma unroll
            for (int q = 0; q < 8; ++q) {
                long long rowg = rowBase + wm * 32 + i * 16 + rsel + q;
                if (rowg < n) {
                    float v = acc[i][j][q] + bv;
                    if (do_relu) v = fmaxf(v, 0.0f);
                    Y[rowg * dout + colg] = v;
                }
            }
        }
    }
}

// ---------------- batch-norm ----------------

__global__ void k_bn_stats(const float* __restrict__ y, float* __restrict__ stats, int d) {
    int col = blockIdx.x * blockDim.x + threadIdx.x;
    if (col >= d) return;
    float s = 0.0f, s2 = 0.0f;
    for (int r = blockIdx.y; r < NN; r += gridDim.y) {
        float v = y[(long long)r * d + col];
        s += v;
        s2 += v * v;
    }
    atomicAdd(&stats[col], s);
    atomicAdd(&stats[d + col], s2);
}

__global__ void k_bn_apply(const float* __restrict__ y, const float* __restrict__ stats,
                           const float* __restrict__ g, const float* __restrict__ be,
                           unsigned short* __restrict__ hout, int d) {
    long long i = (long long)blockIdx.x * blockDim.x + threadIdx.x;
    long long n = (long long)NN * d;
    if (i >= n) return;
    int col = (int)(i % d);
    float mu  = stats[col] * (1.0f / NN);
    float var = stats[d + col] * (1.0f / NN) - mu * mu;
    float v = (y[i] - mu) * rsqrtf(var + EPS) * g[col] + be[col];
    hout[i] = f2bf(v);
}

// ---------------- log_softmax over d=40 cols, one wave32 per row ----------------

__global__ __launch_bounds__(256) void k_logsoftmax(const float* __restrict__ y,
                                                    float* __restrict__ out, int d) {
    int row  = blockIdx.x * (blockDim.x >> 5) + (threadIdx.x >> 5);
    int lane = threadIdx.x & 31;
    if (row >= NN) return;
    const float* p = y + (long long)row * d;
    float v0 = (lane < d)      ? p[lane]      : -INFINITY;
    float v1 = (lane + 32 < d) ? p[lane + 32] : -INFINITY;
    float m = fmaxf(v0, v1);
#pragma unroll
    for (int off = 16; off; off >>= 1) m = fmaxf(m, __shfl_xor(m, off, 32));
    float s = ((lane < d) ? expf(v0 - m) : 0.0f) + ((lane + 32 < d) ? expf(v1 - m) : 0.0f);
#pragma unroll
    for (int off = 16; off; off >>= 1) s += __shfl_xor(s, off, 32);
    float ls = logf(s);
    if (lane < d)      out[(long long)row * d + lane]      = v0 - m - ls;
    if (lane + 32 < d) out[(long long)row * d + lane + 32] = v1 - m - ls;
}

// ---------------- host orchestration ----------------

extern "C" void kernel_launch(void* const* d_in, const int* in_sizes, int n_in,
                              void* d_out, int out_size, void* d_ws, size_t ws_size,
                              hipStream_t stream) {
    (void)in_sizes; (void)n_in; (void)out_size; (void)ws_size;
    const int dims[6] = {128, 128, 256, 512, 1024, 40};

    const float* x = (const float*)d_in[0];
    const long long* edges = (const long long*)d_in[1];
    const long long* src = edges;
    const long long* dstp = edges + NE;

    // workspace layout
    size_t off = 0;
    auto alloc = [&](size_t bytes) -> char* {
        char* p = (char*)d_ws + off;
        off += (bytes + 255) & ~(size_t)255;
        return p;
    };
    float* deg           = (float*)alloc((size_t)NN * 4);
    float* agg           = (float*)alloc((size_t)NN * 1024 * 4);
    unsigned short* aggb = (unsigned short*)alloc((size_t)NN * 1024 * 2);
    unsigned short* h0   = (unsigned short*)alloc((size_t)NN * 1024 * 2);
    unsigned short* h1   = (unsigned short*)alloc((size_t)NN * 1024 * 2);
    float* y             = (float*)alloc((size_t)NN * 1024 * 4);
    unsigned short* Wlb  = (unsigned short*)alloc((size_t)1024 * 1024 * 2);
    unsigned short* Wrb  = (unsigned short*)alloc((size_t)1024 * 1024 * 2);
    float* stats         = (float*)alloc((size_t)2 * 1024 * 4);

    auto cdiv = [](long long a, long long b) { return (int)((a + b - 1) / b); };

    // degree (graph is fixed across layers)
    k_zero<<<512, 256, 0, stream>>>(deg, (long long)NN);
    k_degree<<<cdiv(NE, 256), 256, 0, stream>>>(dstp, deg);

    // x -> bf16
    k_cvt_f2bf<<<cdiv((long long)NN * 128, 256), 256, 0, stream>>>(x, h0, (long long)NN * 128);

    unsigned short* hcur = h0;
    unsigned short* hnext = h1;

    for (int L = 0; L < 5; ++L) {
        int din = dims[L], dout = dims[L + 1];
        const float* Wl = (const float*)d_in[2 + 3 * L];
        const float* Wr = (const float*)d_in[3 + 3 * L];
        const float* b  = (const float*)d_in[4 + 3 * L];

        long long wn = (long long)din * dout;
        k_cvt_w<<<cdiv(wn, 256), 256, 0, stream>>>(Wl, Wlb, din, dout);
        k_cvt_w<<<cdiv(wn, 256), 256, 0, stream>>>(Wr, Wrb, din, dout);

        // mean aggregation
        k_zero<<<2048, 256, 0, stream>>>(agg, (long long)NN * din);
        k_scatter<<<cdiv((long long)NE * (din / 8), 256), 256, 0, stream>>>(hcur, src, dstp, agg, din);
        k_norm<<<cdiv((long long)NN * din, 256), 256, 0, stream>>>(agg, deg, aggb, din);

        // y = agg@Wl + h@Wr + b  (+ReLU for layers 0..3)
        dim3 gg(cdiv(NN, 128), cdiv(dout, 128));
        k_gemm<<<gg, 256, 0, stream>>>(aggb, Wlb, hcur, Wrb, b, y, NN, din, dout, L < 4 ? 1 : 0);

        if (L < 4) {
            const float* g  = (const float*)d_in[17 + 2 * L];
            const float* be = (const float*)d_in[18 + 2 * L];
            k_zero<<<64, 256, 0, stream>>>(stats, (long long)2 * dout);
            dim3 gs(cdiv(dout, 256), 256);
            k_bn_stats<<<gs, 256, 0, stream>>>(y, stats, dout);
            k_bn_apply<<<cdiv((long long)NN * dout, 256), 256, 0, stream>>>(y, stats, g, be, hnext, dout);
            unsigned short* t = hcur; hcur = hnext; hnext = t;
        }
    }

    // final: log_softmax over 40 classes
    k_logsoftmax<<<cdiv(NN, 8), 256, 0, stream>>>(y, (float*)d_out, 40);
}
